// BracketFunc_30915174596782
// MI455X (gfx1250) — matrix operand order repaired and checked by reading, not compiled, not run on hardware
//
#include <hip/hip_runtime.h>
#include <hip/hip_bf16.h>

typedef __attribute__((ext_vector_type(16))) __bf16 v16bf;
typedef __attribute__((ext_vector_type(8)))  float  v8f;

#define SLEN   512
#define BATCH  128
#define DMODEL 1024
#define NHEAD  8
#define HDIM   128
#define MT_TOTAL ((SLEN * BATCH) / 16)   // 4096 M-tiles (per head)

// ---------------------------------------------------------------------------
// Phase 1: pre[s,b,h,:] = x + x @ W_x[h] + b[h]   (written into d_out)
// One WG = 8 waves; wave w owns output columns [16w, 16w+16) of its head.
// ---------------------------------------------------------------------------
__global__ __launch_bounds__(256)
void bracket_phase1(const float* __restrict__ src, const float* __restrict__ W,
                    const float* __restrict__ bias, float* __restrict__ out) {
    __shared__ float xlds[16 * HDIM];               // 8 KB: one 16x128 f32 x-tile
    const int h    = blockIdx.y;
    const int lane = threadIdx.x & 31;
    const int wave = threadIdx.x >> 5;
    const int lhi  = lane >> 4;                     // 0 / 1 half-wave
    const int lln  = lane & 15;
    const int n0   = wave * 16;

    // W_x = rows [128,256) of W[h]; build bf16 B-fragments once (registers).
    const float* Wx = W + (size_t)h * 2 * HDIM * HDIM + (size_t)HDIM * HDIM;
    v16bf bw[4];
    #pragma unroll
    for (int c = 0; c < 4; ++c) {
        #pragma unroll
        for (int j = 0; j < 8; ++j) {
            const int K = 32 * c + ((j < 4) ? 2 * j : 16 + 2 * (j - 4)) + lhi * 8;        // B 32x16 layout
            bw[c][2 * j]     = (__bf16)Wx[(size_t)K       * HDIM + n0 + lln];
            bw[c][2 * j + 1] = (__bf16)Wx[(size_t)(K + 1) * HDIM + n0 + lln];
        }
    }
    const float bsum = bias[h * HDIM + n0 + lln];   // bias depends only on N

    for (int mt = blockIdx.x; mt < MT_TOTAL; mt += gridDim.x) {
        const size_t row0 = (size_t)mt * 16;
        __syncthreads();                            // previous tile fully read
        #pragma unroll
        for (int i = threadIdx.x; i < 16 * HDIM; i += 256) {
            const int r = i >> 7, cc = i & (HDIM - 1);
            xlds[i] = src[(row0 + r) * DMODEL + h * HDIM + cc];
        }
        __syncthreads();

        v8f acc = {};
        #pragma unroll
        for (int c = 0; c < 4; ++c) {
            v16bf a;
            #pragma unroll
            for (int j = 0; j < 8; ++j) {           // A 16x32 bf16 layout
                const int kb = 32 * c + ((j < 4) ? 2 * j : 16 + 2 * (j - 4)) + lhi * 8;
                a[2 * j]     = (__bf16)xlds[lln * HDIM + kb];
                a[2 * j + 1] = (__bf16)xlds[lln * HDIM + kb + 1];
            }
            acc = __builtin_amdgcn_wmma_f32_16x16x32_bf16(false, a, false, bw[c],
                                                          (short)0, acc, false, false);
        }
        #pragma unroll
        for (int j = 0; j < 8; ++j) {               // C/D f32 layout
            const int M = j + lhi * 8;
            const float v = acc[j] + bsum + xlds[M * HDIM + n0 + lln];
            out[(row0 + M) * DMODEL + h * HDIM + n0 + lln] = v;
        }
    }
}

// ---------------------------------------------------------------------------
// Phase 2: r_t = pre_t + r_{t-1} @ W_c[h], in place over d_out (pre -> r).
// 64 WGs (8 batch tiles x 8 heads); r state = bf16 16x128 tile in LDS.
// ---------------------------------------------------------------------------
__global__ __launch_bounds__(256)
void bracket_phase2(const float* __restrict__ W, float* __restrict__ out) {
    __shared__ __bf16 rbf[16 * HDIM];               // 4 KB running-state tile
    const int h    = blockIdx.y;
    const int b0   = blockIdx.x * 16;
    const int lane = threadIdx.x & 31;
    const int wave = threadIdx.x >> 5;
    const int lhi  = lane >> 4;
    const int lln  = lane & 15;
    const int n0   = wave * 16;

    // W_c = rows [0,128) of W[h]
    const float* Wc = W + (size_t)h * 2 * HDIM * HDIM;
    v16bf bw[4];
    #pragma unroll
    for (int c = 0; c < 4; ++c) {
        #pragma unroll
        for (int j = 0; j < 8; ++j) {
            const int K = 32 * c + ((j < 4) ? 2 * j : 16 + 2 * (j - 4)) + lhi * 8;
            bw[c][2 * j]     = (__bf16)Wc[(size_t)K       * HDIM + n0 + lln];
            bw[c][2 * j + 1] = (__bf16)Wc[(size_t)(K + 1) * HDIM + n0 + lln];
        }
    }

    const size_t col = (size_t)h * HDIM + n0 + lln;

    // t = 0: r_0 = pre_0 (already correct in d_out); seed LDS state.
    #pragma unroll
    for (int j = 0; j < 8; ++j) {
        const int M = j + lhi * 8;
        rbf[M * HDIM + n0 + lln] = (__bf16)out[(size_t)(b0 + M) * DMODEL + col];
    }
    __syncthreads();

    for (int t = 1; t < SLEN; ++t) {
        const size_t base = (size_t)t * BATCH * DMODEL;
        float pre[8];                               // issue pre_t loads early
        #pragma unroll
        for (int j = 0; j < 8; ++j) {
            const int M = j + lhi * 8;
            pre[j] = out[base + (size_t)(b0 + M) * DMODEL + col];
        }

        v8f acc = {};
        #pragma unroll
        for (int c = 0; c < 4; ++c) {
            v16bf a;
            #pragma unroll
            for (int j = 0; j < 8; ++j) {
                const int kb = 32 * c + ((j < 4) ? 2 * j : 16 + 2 * (j - 4)) + lhi * 8;
                a[2 * j]     = rbf[lln * HDIM + kb];
                a[2 * j + 1] = rbf[lln * HDIM + kb + 1];
            }
            acc = __builtin_amdgcn_wmma_f32_16x16x32_bf16(false, a, false, bw[c],
                                                          (short)0, acc, false, false);
        }
        __syncthreads();                            // all waves read r_{t-1}

        #pragma unroll
        for (int j = 0; j < 8; ++j) {
            const int M = j + lhi * 8;
            const float v = pre[j] + acc[j];
            out[base + (size_t)(b0 + M) * DMODEL + col] = v;
            rbf[M * HDIM + n0 + lln] = (__bf16)v;   // publish r_t
        }
        __syncthreads();
    }
}

extern "C" void kernel_launch(void* const* d_in, const int* in_sizes, int n_in,
                              void* d_out, int out_size, void* d_ws, size_t ws_size,
                              hipStream_t stream) {
    (void)in_sizes; (void)n_in; (void)d_ws; (void)ws_size; (void)out_size;
    const float* src = (const float*)d_in[0];
    const float* W   = (const float*)d_in[1];
    const float* b   = (const float*)d_in[2];
    float* out = (float*)d_out;

    dim3 blk(256);
    dim3 g1(1024, NHEAD);
    bracket_phase1<<<g1, blk, 0, stream>>>(src, W, b, out);
    dim3 g2(BATCH / 16, NHEAD);
    bracket_phase2<<<g2, blk, 0, stream>>>(W, out);
}
